// MultiHeadAttention_21921513079143
// MI455X (gfx1250) — matrix operand order repaired and checked by reading, not compiled, not run on hardware
//
#include <hip/hip_runtime.h>
#include <hip/hip_bf16.h>

// ---------------- problem constants ----------------
#define NB 16
#define NA 512
#define HID 256
#define NHD 8
#define DH 32          // head dim
#define INV_TP 0.125f  // 1/sqrt(2*DH) = 1/8

typedef __attribute__((ext_vector_type(16))) __bf16 v16bf;
typedef __attribute__((ext_vector_type(8)))  float  v8f;

union BFrag {
    v16bf v;
    unsigned short u[16];
    unsigned int   w[8];
};

__device__ __forceinline__ unsigned short f2bf(float f) {
    unsigned int x = __float_as_uint(f);
    unsigned int r = x + 0x7FFFu + ((x >> 16) & 1u);   // RNE truncation to bf16
    return (unsigned short)(r >> 16);
}

__device__ __forceinline__ unsigned int pack_bf2(float a, float b) {
    return (unsigned int)f2bf(a) | ((unsigned int)f2bf(b) << 16);
}

// A-matrix (16x32 bf16) lane layout: lane = half*16 + m; per packed-pair index i (0..7),
// the two K indices start at:
__device__ __forceinline__ int a_kk(int i, int half) {
    return (i < 4) ? (8 * half + 2 * i) : (16 + 8 * half + 2 * (i - 4));
}

// ---------------------------------------------------------------------------
// Prep A: x fp32 -> bf16 (packed pairs), one pass.
// ---------------------------------------------------------------------------
__global__ void cvt_x_bf16(const float* __restrict__ x, unsigned int* __restrict__ xbf) {
    int i = blockIdx.x * blockDim.x + threadIdx.x;      // pair index, 1,048,576 total
    float2 f = reinterpret_cast<const float2*>(x)[i];
    xbf[i] = pack_bf2(f.x, f.y);
}

// ---------------------------------------------------------------------------
// Prep B: W (HIDxHID, row-major) -> Wt bf16 with Wt[k][n] = W[n][k], for q/k/v.
// Stage-1 B-fragment 16-bit loads are then contiguous across lanes (n = m16).
// ---------------------------------------------------------------------------
__global__ void cvt_w_bf16t(const float* __restrict__ Wq,
                            const float* __restrict__ Wk,
                            const float* __restrict__ Wv,
                            unsigned int* __restrict__ wt) {
    int gid = blockIdx.x * blockDim.x + threadIdx.x;    // 3*256*128 = 98304 total
    int pn   = gid & 127;         // n-pair
    int k    = (gid >> 7) & 255;  // k
    int proj = gid >> 15;         // 0..2
    const float* W = (proj == 0) ? Wq : ((proj == 1) ? Wk : Wv);
    float a = W[(2 * pn)     * HID + k];
    float b = W[(2 * pn + 1) * HID + k];
    wt[proj * (HID * HID / 2) + k * (HID / 2) + pn] = pack_bf2(a, b);
}

// ---------------------------------------------------------------------------
// Kernel 1: QKV projection, pure bf16 loads + WMMA.
//   q -> [b][h][n][d]   k -> [b][h][d][n] (transposed!)   v -> [b][h][n][d]
// One wave per 16x16 output tile; 8 x v_wmma_f32_16x16x32_bf16 over K=256.
// ---------------------------------------------------------------------------
__global__ void mha_qkv_proj(const unsigned int* __restrict__ xbf,
                             const unsigned int* __restrict__ wt,
                             unsigned short* __restrict__ q_ws,
                             unsigned short* __restrict__ kT_ws,
                             unsigned short* __restrict__ v_ws) {
    int wid  = (blockIdx.x * blockDim.x + threadIdx.x) >> 5;
    int lane = threadIdx.x & 31;

    int ct = wid & 15;  wid >>= 4;   // 16 column tiles
    int rt = wid & 31;  wid >>= 5;   // 32 row tiles
    int b  = wid & 15;  wid >>= 4;   // 16 batches
    int proj = wid;                  // 0=q, 1=k, 2=v

    int m16  = lane & 15;
    int half = lane >> 4;

    v8f acc = {0.f, 0.f, 0.f, 0.f, 0.f, 0.f, 0.f, 0.f};

    const unsigned int* xrow32 = xbf + (size_t)(b * NA + rt * 16 + m16) * (HID / 2);
    const unsigned int* wbase  = wt + proj * (HID * HID / 2) + ct * 8;

    #pragma unroll
    for (int kb = 0; kb < HID; kb += 32) {
        BFrag a, bw;
        #pragma unroll
        for (int i = 0; i < 8; ++i)
            a.w[i] = xrow32[(kb + a_kk(i, half)) >> 1];
        const unsigned short* wcol = reinterpret_cast<const unsigned short*>(wbase) + m16;
        #pragma unroll
        for (int i = 0; i < 8; ++i) {
            bw.u[2 * i]     = wcol[(kb + a_kk(i, half)) * HID];
            bw.u[2 * i + 1] = wcol[(kb + a_kk(i, half) + 1) * HID];
        }
        acc = __builtin_amdgcn_wmma_f32_16x16x32_bf16(false, a.v, false, bw.v,
                                                      (short)0, acc, false, false);
    }

    // D layout: M = r + 8*half, N = m16
    #pragma unroll
    for (int r = 0; r < 8; ++r) {
        int m   = r + 8 * half;
        int j   = ct * 16 + m16;       // global output column
        int h   = j >> 5;
        int d   = j & 31;
        int row = rt * 16 + m;
        unsigned short val = f2bf(acc[r]);
        if (proj == 0)
            q_ws[(((b * NHD + h) * NA) + row) * DH + d] = val;
        else if (proj == 1)
            kT_ws[(((b * NHD + h) * DH) + d) * NA + row] = val;   // transposed
        else
            v_ws[(((b * NHD + h) * NA) + row) * DH + d] = val;
    }
}

// ---------------------------------------------------------------------------
// Kernel 2: fused flash-style attention.  One wave per (b, head, 16-query tile).
// 16 chunks of 32 keys: 2 score WMMAs + online softmax + 2 P*V WMMAs.
// ---------------------------------------------------------------------------
__global__ void mha_attention(const float* __restrict__ x,
                              const unsigned char* __restrict__ mask,
                              const float* __restrict__ adj,
                              const unsigned short* __restrict__ q_ws,
                              const unsigned short* __restrict__ kT_ws,
                              const unsigned short* __restrict__ v_ws,
                              float* __restrict__ out) {
    __shared__ float          sS[8][16][32];
    __shared__ unsigned short sP[8][16][32];
    __shared__ float          sCorr[8][16];

    int wv   = threadIdx.x >> 5;
    int lane = threadIdx.x & 31;
    int t    = blockIdx.x * (blockDim.x >> 5) + wv;

    int qt = t & 31;  t >>= 5;   // 32 query tiles
    int h  = t & 7;   t >>= 3;   // 8 heads
    int b  = t;                  // 16 batches

    int m16  = lane & 15;
    int half = lane >> 4;
    int qbase = qt * 16;

    const unsigned short* qb_ptr = q_ws  + ((size_t)(b * NHD + h) * NA) * DH;
    const unsigned short* kt_ptr = kT_ws + ((size_t)(b * NHD + h) * DH) * NA;
    const unsigned short* vb_ptr = v_ws  + ((size_t)(b * NHD + h) * NA) * DH;

    // Load Q fragment once (K-dim = DH = 32 -> exactly one WMMA step)
    BFrag qa;
    {
        const unsigned short* qrow = qb_ptr + (qbase + m16) * DH;
        #pragma unroll
        for (int i = 0; i < 8; ++i)
            qa.w[i] = *reinterpret_cast<const unsigned int*>(qrow + a_kk(i, half));
    }

    // This lane's key column (n = m16) in K^T; elements stride over d
    const unsigned short* ktrow = kt_ptr + m16;

    v8f o0 = {0.f, 0.f, 0.f, 0.f, 0.f, 0.f, 0.f, 0.f};
    v8f o1 = {0.f, 0.f, 0.f, 0.f, 0.f, 0.f, 0.f, 0.f};
    float runm = -1e30f;   // running row max (lanes 0..15)
    float runl = 0.0f;     // running row sum (lanes 0..15)

    for (int kc = 0; kc < 16; ++kc) {
        int kb = kc * 32;

        // Prefetch next chunk's K/V rows (lowers to global_prefetch_b8)
        if (kc < 15) {
            __builtin_prefetch(ktrow + kb + 32, 0, 1);
            __builtin_prefetch(vb_ptr + (kb + 32 + lane) * DH, 0, 1);
        }

        // --- S = Q * K^T : B fragments, 16-bit loads, contiguous across lanes along K^T rows ---
        BFrag bk0, bk1;
        #pragma unroll
        for (int i = 0; i < 8; ++i) {
            int d0 = a_kk(i, half);
            bk0.u[2 * i]     = ktrow[d0 * NA + kb];
            bk0.u[2 * i + 1] = ktrow[(d0 + 1) * NA + kb];
            bk1.u[2 * i]     = ktrow[d0 * NA + kb + 16];
            bk1.u[2 * i + 1] = ktrow[(d0 + 1) * NA + kb + 16];
        }
        v8f zc = {0.f, 0.f, 0.f, 0.f, 0.f, 0.f, 0.f, 0.f};
        v8f s0 = __builtin_amdgcn_wmma_f32_16x16x32_bf16(false, qa.v, false, bk0.v,
                                                         (short)0, zc, false, false);
        v8f s1 = __builtin_amdgcn_wmma_f32_16x16x32_bf16(false, qa.v, false, bk1.v,
                                                         (short)0, zc, false, false);

        // --- scale + adjacency + mask, spill to LDS for row softmax ---
        #pragma unroll
        for (int r = 0; r < 8; ++r) {
            int m   = r + 8 * half;
            int row = qbase + m;
            int base = (b * NA + row) * NA;
            int c0 = kb + m16;
            int c1 = kb + 16 + m16;
            float v0 = s0[r] * INV_TP + adj[base + c0];
            float v1 = s1[r] * INV_TP + adj[base + c1];
            if (mask[base + c0]) v0 = -1e30f;
            if (mask[base + c1]) v1 = -1e30f;
            sS[wv][m][m16]      = v0;
            sS[wv][m][16 + m16] = v1;
        }
        __syncthreads();

        // --- online softmax: lanes 0..15 each own one row ---
        if (lane < 16) {
            float sv[32];
            float mx = -1e30f;
            #pragma unroll
            for (int j = 0; j < 32; ++j) {
                sv[j] = sS[wv][lane][j];
                mx = fmaxf(mx, sv[j]);
            }
            float newm = fmaxf(runm, mx);
            float corr = __expf(runm - newm);
            float sum = 0.f;
            #pragma unroll
            for (int j = 0; j < 32; ++j) {
                float p = __expf(sv[j] - newm);
                sum += p;
                sP[wv][lane][j] = f2bf(p);
            }
            runl = runl * corr + sum;
            runm = newm;
            sCorr[wv][lane] = corr;
        }
        __syncthreads();

        // --- rescale accumulators by correction factor ---
        #pragma unroll
        for (int r = 0; r < 8; ++r) {
            float c = sCorr[wv][r + 8 * half];
            o0[r] *= c;
            o1[r] *= c;
        }

        // --- O += P * V : A from LDS P, B from V columns (K-dim = keys over elements) ---
        BFrag pa;
        {
            const unsigned short* prow = &sP[wv][m16][0];
            #pragma unroll
            for (int i = 0; i < 8; ++i)
                pa.w[i] = *reinterpret_cast<const unsigned int*>(prow + a_kk(i, half));
        }
        BFrag bv0, bv1;
        {
            const unsigned short* vcol = vb_ptr + kb * DH + m16;
            #pragma unroll
            for (int i = 0; i < 8; ++i) {
                int k0 = a_kk(i, half);
                bv0.u[2 * i]     = vcol[k0 * DH];
                bv0.u[2 * i + 1] = vcol[(k0 + 1) * DH];
                bv1.u[2 * i]     = vcol[k0 * DH + 16];
                bv1.u[2 * i + 1] = vcol[(k0 + 1) * DH + 16];
            }
        }
        o0 = __builtin_amdgcn_wmma_f32_16x16x32_bf16(false, pa.v, false, bv0.v,
                                                     (short)0, o0, false, false);
        o1 = __builtin_amdgcn_wmma_f32_16x16x32_bf16(false, pa.v, false, bv1.v,
                                                     (short)0, o1, false, false);
        __syncthreads();
    }

    // --- normalize, residual add, write out ---
    if (lane < 16) sCorr[wv][lane] = 1.0f / fmaxf(runl, 1e-30f);
    __syncthreads();

    #pragma unroll
    for (int r = 0; r < 8; ++r) {
        int m   = r + 8 * half;
        float inv = sCorr[wv][m];
        int row = qbase + m;
        int ob  = (b * NA + row) * HID + h * DH;
        out[ob + m16]      = o0[r] * inv + x[ob + m16];
        out[ob + 16 + m16] = o1[r] * inv + x[ob + 16 + m16];
    }
}

// ---------------------------------------------------------------------------
extern "C" void kernel_launch(void* const* d_in, const int* in_sizes, int n_in,
                              void* d_out, int out_size, void* d_ws, size_t ws_size,
                              hipStream_t stream) {
    (void)in_sizes; (void)n_in; (void)out_size; (void)ws_size;

    const float*         x    = (const float*)d_in[0];
    const unsigned char* mask = (const unsigned char*)d_in[1];
    const float*         adj  = (const float*)d_in[2];
    const float*         Wq   = (const float*)d_in[3];
    const float*         Wk   = (const float*)d_in[4];
    const float*         Wv   = (const float*)d_in[5];
    float*               out  = (float*)d_out;

    // workspace layout (bf16 elements)
    const size_t n_x  = (size_t)NB * NA * HID;           // 2,097,152
    const size_t n_w  = (size_t)3 * HID * HID;           //   196,608
    const size_t n_p  = (size_t)NB * NHD * NA * DH;      // 2,097,152 per tensor
    unsigned short* xbf   = (unsigned short*)d_ws;
    unsigned short* wtbf  = xbf  + n_x;
    unsigned short* q_ws  = wtbf + n_w;
    unsigned short* kT_ws = q_ws + n_p;
    unsigned short* v_ws  = kT_ws + n_p;

    // Prep: one-pass bf16 conversion (x) and transposed bf16 weights
    cvt_x_bf16<<<(int)(n_x / 2 / 256), 256, 0, stream>>>(x, (unsigned int*)xbf);
    cvt_w_bf16t<<<(3 * HID * HID / 2) / 256, 256, 0, stream>>>(Wq, Wk, Wv,
                                                               (unsigned int*)wtbf);

    // Stage 1: 3 proj * 16 b * 32 rowtiles * 16 coltiles = 24576 waves, 8 waves/block
    mha_qkv_proj<<<24576 / 8, 256, 0, stream>>>((const unsigned int*)xbf,
                                                (const unsigned int*)wtbf,
                                                q_ws, kT_ws, v_ws);

    // Stage 2: 16 b * 8 heads * 32 qtiles = 4096 waves, 8 waves/block
    mha_attention<<<4096 / 8, 256, 0, stream>>>(x, mask, adj, q_ws, kT_ws, v_ws, out);
}
